// AttModule_mamba_1116691497196
// MI455X (gfx1250) — matrix operand order repaired and checked by reading, not compiled, not run on hardware
//
#include <hip/hip_runtime.h>
#include <hip/hip_bf16.h>
#include <math.h>

// Problem constants (match reference)
#define B_    4
#define C_    256
#define L_    2048
#define DI_   512
#define DS_   16
#define ROWS_ (B_ * L_)      // 8192
#define EPS_  1e-5f

typedef unsigned int u32;
typedef __attribute__((ext_vector_type(16))) __bf16 v16bf;
typedef __attribute__((ext_vector_type(8)))  float  v8f;
typedef __attribute__((ext_vector_type(4)))  u32    u32x4;
typedef __attribute__((ext_vector_type(8)))  int    i32x8;
typedef __attribute__((ext_vector_type(4)))  int    i32x4;

// ---------------------------------------------------------------------------
// Tensor Data Mover: 2-D f32 tile (tile_rows x width) from global -> LDS.
// D# packed per CDNA5 ISA 8.3/8.4: group0 {count=1, lds_addr, global_addr,
// type=2}; group1 {data_size=2 (4B), tensor_dim0=width, tensor_dim1=rows,
// tile_dim0=width, tile_dim1=tile_rows, tensor_dim0_stride=width}.
// ---------------------------------------------------------------------------
__device__ __forceinline__ void tdm_load_2d_f32(u32 lds_off, const void* gaddr,
                                                u32 width, u32 rows_total,
                                                u32 tile_rows) {
    unsigned long long ga = (unsigned long long)gaddr;
    u32x4 g0;
    g0[0] = 1u;                                        // count=1 (valid), user
    g0[1] = lds_off;                                   // lds_addr (bytes)
    g0[2] = (u32)(ga & 0xFFFFFFFFu);                   // global_addr[31:0]
    g0[3] = (u32)((ga >> 32) & 0x1FFFFFFu) | (2u << 30); // addr[56:32] | type=2
    i32x8 g1;
    g1[0] = (int)(2u << 16);                           // wg_mask=0, data_size=4B
    g1[1] = (int)((width & 0xFFFFu) << 16);            // tensor_dim0[15:0]
    g1[2] = (int)(((width >> 16) & 0xFFFFu) |
                  ((rows_total & 0xFFFFu) << 16));     // td0[31:16] | td1[15:0]
    g1[3] = (int)(((rows_total >> 16) & 0xFFFFu) |
                  ((width & 0xFFFFu) << 16));          // td1[31:16] | tile_dim0
    g1[4] = (int)(tile_rows & 0xFFFFu);                // tile_dim1, tile_dim2=0
    g1[5] = (int)width;                                // tensor_dim0_stride[31:0]
    g1[6] = 0;                                         // stride[47:32], td1_stride
    g1[7] = 0;
    i32x4 z4 = (i32x4)0;
#if defined(__clang_major__) && (__clang_major__ >= 23)
    i32x8 z8 = (i32x8)0;
    __builtin_amdgcn_tensor_load_to_lds(g0, g1, z4, z4, z8, 0);
#else
    __builtin_amdgcn_tensor_load_to_lds(g0, g1, z4, z4, 0);
#endif
}

// Weight f32 -> bf16 conversion staged through LDS via the TDM.
#define WCONV_ROWS 16
__global__ void k_wconv_bf16(const float* __restrict__ W, __bf16* __restrict__ Wb,
                             int Kdim, int rows_total) {
    __shared__ float tile[WCONV_ROWS * 512];           // <= 32 KB
    const int r0 = blockIdx.x * WCONV_ROWS;
    if (threadIdx.x < 32) {                            // wave 0 issues the DMA
        tdm_load_2d_f32(0u, W + (size_t)r0 * Kdim, (u32)Kdim,
                        (u32)rows_total, (u32)WCONV_ROWS);
        __builtin_amdgcn_s_wait_tensorcnt(0);
    }
    __syncthreads();
    const int total = WCONV_ROWS * Kdim;
    for (int i = threadIdx.x; i < total; i += 256)
        Wb[(size_t)r0 * Kdim + i] = (__bf16)tile[i];
}

// ---------------------------------------------------------------------------
// WMMA fragment loaders (per CDNA5 ISA 7.12.2 layouts, wave32)
// A 16x32 bf16: lane<16 -> row=lane,    K = k0..k0+7 | k0+16..k0+23
//               lane>=16 -> row=lane-16, K = k0+8..15 | k0+24..31
// ---------------------------------------------------------------------------
__device__ __forceinline__ v16bf load_a_frag(const __bf16* A, int lda, int m0,
                                             int k0, int lane) {
    int row = m0 + (lane & 15);
    int kb  = k0 + ((lane & 16) ? 8 : 0);
    union { uint4 u[2]; v16bf v; } r;
    const __bf16* p = A + (size_t)row * lda + kb;
    r.u[0] = *(const uint4*)(p);        // 8 bf16 = 16B
    r.u[1] = *(const uint4*)(p + 16);   // next 8 bf16
    return r.v;
}

// B 32x16 (KxN) from bf16 weights stored (N,K) row-major: lane<16: n=lane,
// K=k0..k0+7 | k0+16..k0+23; lane>=16: n=lane-16, K=k0+8..15 | k0+24..31.
__device__ __forceinline__ v16bf load_b_frag(const __bf16* Wb, int ldw, int n0,
                                             int k0, int lane) {
    int nn = n0 + (lane & 15);
    int kb = k0 + ((lane & 16) ? 8 : 0);
    union { uint4 u[2]; v16bf v; } r;
    const __bf16* p = Wb + (size_t)nn * ldw + kb;
    r.u[0] = *(const uint4*)(p);        // K = kb..kb+7
    r.u[1] = *(const uint4*)(p + 16);   // K = kb+16..kb+23
    return r.v;
}

// Wave-level WMMA GEMM: C(M,N) = A(M,K)bf16 * Wb(N,K)bf16^T, f32 accumulate.
// Each wave owns a 32 x (16*NT) tile (2 M-tiles reuse each B fragment).
// Grid supplies exactly (M/32)*(N/(16*NT)) waves; control flow wave-uniform.
template <int NT>
__global__ void k_gemm_bf16(const __bf16* __restrict__ A,
                            const __bf16* __restrict__ Wb,
                            float* __restrict__ Cout, int M, int N, int K) {
    int gtid  = blockIdx.x * blockDim.x + threadIdx.x;
    int wave  = gtid >> 5;
    int lane  = threadIdx.x & 31;
    int wavesN = N / (16 * NT);
    int m0 = (wave / wavesN) * 32;
    int n0 = (wave % wavesN) * (16 * NT);
    if (m0 >= M) return;  // wave-uniform

    v8f acc0[NT], acc1[NT];
#pragma unroll
    for (int t = 0; t < NT; ++t) { acc0[t] = (v8f){}; acc1[t] = (v8f){}; }

    for (int k0 = 0; k0 < K; k0 += 32) {
        if (k0 + 32 < K)  // prefetch next A K-tile -> global_prefetch_b8
            __builtin_prefetch(A + (size_t)(m0 + (lane & 15)) * K + k0 + 32, 0, 1);
        v16bf a0 = load_a_frag(A, K, m0,      k0, lane);
        v16bf a1 = load_a_frag(A, K, m0 + 16, k0, lane);
#pragma unroll
        for (int t = 0; t < NT; ++t) {
            v16bf b = load_b_frag(Wb, K, n0 + 16 * t, k0, lane);
            acc0[t] = __builtin_amdgcn_wmma_f32_16x16x32_bf16(
                false, a0, false, b, (short)0, acc0[t], false, false);
            acc1[t] = __builtin_amdgcn_wmma_f32_16x16x32_bf16(
                false, a1, false, b, (short)0, acc1[t], false, false);
        }
    }
    // D layout: VGPR r -> row +r (lanes 0-15) / +8+r (lanes 16-31), col=lane&15
    int rb0  = m0 + ((lane & 16) ? 8 : 0);
    int col0 = (lane & 15);
#pragma unroll
    for (int t = 0; t < NT; ++t) {
        int n = n0 + 16 * t + col0;
#pragma unroll
        for (int r = 0; r < 8; ++r) {
            Cout[(size_t)(rb0 + r) * N + n]      = acc0[t][r];
            Cout[(size_t)(rb0 + 16 + r) * N + n] = acc1[t][r];
        }
    }
}

// ---------------------------------------------------------------------------
// 1) dilated conv1d (k=3, dil=2, pad=2) + bias + ReLU + instance-norm over L
// ---------------------------------------------------------------------------
__global__ void k_ffconv_inorm(const float* __restrict__ x,
                               const float* __restrict__ ff_w,
                               const float* __restrict__ ff_b,
                               float* __restrict__ conv_out,
                               float* __restrict__ inorm) {
    const int bc = blockIdx.x;        // b*C + c
    const int b  = bc >> 8;
    const int c  = bc & (C_ - 1);
    const int t  = threadIdx.x;       // 256 threads

    __shared__ float sx[L_ + 4];
    __shared__ float r1[256], r2[256];

    if (t < 2) { sx[t] = 0.f; sx[L_ + 2 + t] = 0.f; }

    float acc[8];
#pragma unroll
    for (int j = 0; j < 8; ++j) acc[j] = 0.f;

    const float* xb = x + (size_t)b * C_ * L_;
    for (int ci = 0; ci < C_; ++ci) {
        __syncthreads();
        for (int i = t; i < L_; i += 256) sx[i + 2] = xb[(size_t)ci * L_ + i];
        __syncthreads();
        const float w0 = ff_w[(c * C_ + ci) * 3 + 0];
        const float w1 = ff_w[(c * C_ + ci) * 3 + 1];
        const float w2 = ff_w[(c * C_ + ci) * 3 + 2];
#pragma unroll
        for (int j = 0; j < 8; ++j) {
            int l = j * 256 + t;
            acc[j] += sx[l] * w0 + sx[l + 2] * w1 + sx[l + 4] * w2;
        }
    }

    const float bias = ff_b[c];
    float s1 = 0.f, s2 = 0.f;
    float v[8];
#pragma unroll
    for (int j = 0; j < 8; ++j) {
        float o = fmaxf(acc[j] + bias, 0.f);
        v[j] = o; s1 += o; s2 += o * o;
        conv_out[(size_t)bc * L_ + j * 256 + t] = o;
    }
    r1[t] = s1; r2[t] = s2; __syncthreads();
    for (int off = 128; off > 0; off >>= 1) {
        if (t < off) { r1[t] += r1[t + off]; r2[t] += r2[t + off]; }
        __syncthreads();
    }
    const float mean = r1[0] * (1.f / L_);
    const float var  = r2[0] * (1.f / L_) - mean * mean;
    const float rstd = rsqrtf(var + EPS_);
#pragma unroll
    for (int j = 0; j < 8; ++j)
        inorm[(size_t)bc * L_ + j * 256 + t] = (v[j] - mean) * rstd;
}

// ---------------------------------------------------------------------------
// 2) LayerNorm over C per (b,l); emit bf16 rows (b*L+l, c) for WMMA
// ---------------------------------------------------------------------------
__global__ void k_layernorm(const float* __restrict__ inorm,
                            const float* __restrict__ g,
                            const float* __restrict__ be,
                            __bf16* __restrict__ h) {
    const int bl = blockIdx.x;              // b*L + l
    const int b  = bl >> 11;
    const int l  = bl & (L_ - 1);
    const int c  = threadIdx.x;
    __shared__ float r1[256], r2[256];
    float v = inorm[((size_t)b * C_ + c) * L_ + l];
    r1[c] = v; r2[c] = v * v; __syncthreads();
    for (int off = 128; off > 0; off >>= 1) {
        if (c < off) { r1[c] += r1[c + off]; r2[c] += r2[c + off]; }
        __syncthreads();
    }
    const float m   = r1[0] * (1.f / C_);
    const float var = r2[0] * (1.f / C_) - m * m;
    const float hn  = (v - m) * rsqrtf(var + EPS_) * g[c] + be[c];
    h[(size_t)bl * C_ + c] = (__bf16)hn;
}

// ---------------------------------------------------------------------------
// 4) depthwise causal conv (k=4) + SiLU; xz rows are (row,1024)
// ---------------------------------------------------------------------------
__global__ void k_dwconv_silu(const float* __restrict__ xz,
                              const float* __restrict__ cw,
                              const float* __restrict__ cb,
                              float* __restrict__ xc,
                              __bf16* __restrict__ xcb) {
    const int idx  = blockIdx.x * 256 + threadIdx.x;   // ROWS_*DI_
    const int d    = idx & (DI_ - 1);
    const int row  = idx >> 9;
    const int l    = row & (L_ - 1);
    const int base = row - l;                          // b*L
    float a = cb[d];
#pragma unroll
    for (int k = 0; k < 4; ++k) {
        int ls = l - 3 + k;
        if (ls >= 0) a += xz[(size_t)(base + ls) * 1024 + d] * cw[d * 4 + k];
    }
    const float s = a / (1.f + expf(-a));  // silu
    xc[idx]  = s;
    xcb[idx] = (__bf16)s;
}

// ---------------------------------------------------------------------------
// 6) dt_proj (K=16, VALU) + softplus
// ---------------------------------------------------------------------------
__global__ void k_dtproj(const float* __restrict__ dbl,
                         const float* __restrict__ dtw,
                         const float* __restrict__ dtb,
                         float* __restrict__ dt) {
    const int idx = blockIdx.x * 256 + threadIdx.x;    // ROWS_*DI_
    const int d   = idx & (DI_ - 1);
    const int row = idx >> 9;
    float s = dtb[d];
    const float* r = dbl + (size_t)row * 48;
#pragma unroll
    for (int j = 0; j < 16; ++j) s += r[j] * dtw[d * 16 + j];
    dt[idx] = (s > 20.f) ? s : log1pf(expf(s));
}

// ---------------------------------------------------------------------------
// 7) selective scan: lane per (b,d,n); 16-lane shuffle tree, SiLU(z) gating
// ---------------------------------------------------------------------------
__global__ void k_scan(const float* __restrict__ dt,
                       const float* __restrict__ xc,
                       const float* __restrict__ dbl,
                       const float* __restrict__ xz,
                       const float* __restrict__ A_log,
                       const float* __restrict__ Dskip,
                       __bf16* __restrict__ y) {
    const int t  = blockIdx.x * 256 + threadIdx.x;     // B_*DI_*DS_ = 32768
    const int n  = t & 15;
    const int ch = t >> 4;
    const int d  = ch & (DI_ - 1);
    const int b  = ch >> 9;
    const float Av = -expf(A_log[d * 16 + n]);
    const float Dv = Dskip[d];
    const int rowbase = b << 11;
    float h = 0.f;
    for (int l = 0; l < L_; ++l) {
        const int row = rowbase + l;
        const float dtv = dt[(size_t)row * DI_ + d];
        const float xv  = xc[(size_t)row * DI_ + d];
        const float Bv  = dbl[(size_t)row * 48 + 16 + n];
        const float Cv  = dbl[(size_t)row * 48 + 32 + n];
        h = expf(dtv * Av) * h + (dtv * xv) * Bv;
        float yv = h * Cv;
        yv += __shfl_xor(yv, 8, 16);
        yv += __shfl_xor(yv, 4, 16);
        yv += __shfl_xor(yv, 2, 16);
        yv += __shfl_xor(yv, 1, 16);
        if (n == 0) {
            const float z  = xz[(size_t)row * 1024 + DI_ + d];
            const float sz = z / (1.f + expf(-z));
            y[(size_t)row * DI_ + d] = (__bf16)((yv + Dv * xv) * sz);
        }
    }
}

// ---------------------------------------------------------------------------
// 9) final fusion: (x + ALPHA*((inorm + att)*pm) + conv_out) * pm
// ---------------------------------------------------------------------------
__global__ void k_final(const float* __restrict__ x,
                        const float* __restrict__ mask,
                        const float* __restrict__ conv_out,
                        const float* __restrict__ inorm,
                        const float* __restrict__ att,
                        float* __restrict__ out) {
    const int idx = blockIdx.x * 256 + threadIdx.x;    // B_*C_*L_
    const int l   = idx & (L_ - 1);
    const int bc  = idx >> 11;
    const int c   = bc & (C_ - 1);
    const int b   = bc >> 8;
    const float pm = mask[b * L_ + l];
    const int row  = b * L_ + l;
    const float a  = (inorm[idx] + att[(size_t)row * C_ + c]) * pm;
    out[idx] = (x[idx] + 1.0f * a + conv_out[idx]) * pm;
}

// ---------------------------------------------------------------------------
extern "C" void kernel_launch(void* const* d_in, const int* in_sizes, int n_in,
                              void* d_out, int out_size, void* d_ws, size_t ws_size,
                              hipStream_t stream) {
    const float* x        = (const float*)d_in[0];
    // d_in[1] = f (unused by the reference)
    const float* mask     = (const float*)d_in[2];
    const float* ff_w     = (const float*)d_in[3];
    const float* ff_b     = (const float*)d_in[4];
    const float* ln_g     = (const float*)d_in[5];
    const float* ln_b     = (const float*)d_in[6];
    const float* in_proj  = (const float*)d_in[7];   // (1024, 256)
    const float* conv_w   = (const float*)d_in[8];   // (512, 1, 4)
    const float* conv_b   = (const float*)d_in[9];
    const float* x_proj   = (const float*)d_in[10];  // (48, 512)
    const float* dt_w     = (const float*)d_in[11];  // (512, 16)
    const float* dt_b     = (const float*)d_in[12];
    const float* A_log    = (const float*)d_in[13];  // (512, 16)
    const float* D_skip   = (const float*)d_in[14];
    const float* out_proj = (const float*)d_in[15];  // (256, 512)
    float* out = (float*)d_out;

    // workspace layout (all 128B aligned by construction)
    char* w = (char*)d_ws;
    float*  conv_out = (float*)(w + 0);                 //  8 MB (B,C,L)
    float*  inorm    = (float*)(w + 8388608);           //  8 MB (B,C,L)
    __bf16* h_bf16   = (__bf16*)(w + 16777216);         //  4 MB (ROWS,256)
    float*  xz       = (float*)(w + 20971520);          // 32 MB (ROWS,1024)
    float*  xc       = (float*)(w + 54525952);          // 16 MB (ROWS,512)
    __bf16* xc_bf16  = (__bf16*)(w + 71303168);         //  8 MB (ROWS,512)
    float*  dbl      = (float*)(w + 79691776);          //1.5 MB (ROWS,48)
    float*  dt       = (float*)(w + 81264640);          // 16 MB (ROWS,512)
    __bf16* y_bf16   = (__bf16*)(w + 98041856);         //  8 MB (ROWS,512)
    float*  att      = (float*)(w + 106430464);         //  8 MB (ROWS,256)
    __bf16* wb_in    = (__bf16*)(w + 114818944);        // 512 KB (1024,256)
    __bf16* wb_x     = (__bf16*)(w + 115343232);        //  48 KB (48,512)
    __bf16* wb_out   = (__bf16*)(w + 115408768);        // 256 KB (256,512)

    const dim3 blk(256);

    // 0) weight f32->bf16 conversion via Tensor Data Mover (TDM -> LDS -> cvt)
    k_wconv_bf16<<<1024 / WCONV_ROWS, blk, 0, stream>>>(in_proj, wb_in, 256, 1024);
    k_wconv_bf16<<<48 / WCONV_ROWS, blk, 0, stream>>>(x_proj, wb_x, 512, 48);
    k_wconv_bf16<<<256 / WCONV_ROWS, blk, 0, stream>>>(out_proj, wb_out, 512, 256);

    // 1) dilated conv + ReLU + instance norm
    k_ffconv_inorm<<<B_ * C_, blk, 0, stream>>>(x, ff_w, ff_b, conv_out, inorm);

    // 2) layernorm over C -> bf16 activations
    k_layernorm<<<ROWS_, blk, 0, stream>>>(inorm, ln_g, ln_b, h_bf16);

    // 3) in_proj WMMA GEMM: (8192,256)bf16 x (1024,256)^T -> xz (8192,1024)
    {
        int waves = (ROWS_ / 32) * (1024 / 64);          // 4096 waves
        k_gemm_bf16<4><<<waves * 32 / 256, blk, 0, stream>>>(
            h_bf16, wb_in, xz, ROWS_, 1024, C_);
    }

    // 4) depthwise causal conv + SiLU
    k_dwconv_silu<<<(ROWS_ * DI_) / 256, blk, 0, stream>>>(xz, conv_w, conv_b,
                                                           xc, xc_bf16);

    // 5) x_proj WMMA GEMM: (8192,512)bf16 x (48,512)^T -> dbl (8192,48)
    {
        int waves = (ROWS_ / 32) * (48 / 48);            // 256 waves
        k_gemm_bf16<3><<<waves * 32 / 256, blk, 0, stream>>>(
            xc_bf16, wb_x, dbl, ROWS_, 48, DI_);
    }

    // 6) dt_proj + softplus
    k_dtproj<<<(ROWS_ * DI_) / 256, blk, 0, stream>>>(dbl, dt_w, dt_b, dt);

    // 7) selective scan (lane per state element, shuffle reduce) + gating
    k_scan<<<(B_ * DI_ * DS_) / 256, blk, 0, stream>>>(dt, xc, dbl, xz, A_log,
                                                       D_skip, y_bf16);

    // 8) out_proj WMMA GEMM: (8192,512)bf16 x (256,512)^T -> att (8192,256)
    {
        int waves = (ROWS_ / 32) * (256 / 64);           // 1024 waves
        k_gemm_bf16<4><<<waves * 32 / 256, blk, 0, stream>>>(
            y_bf16, wb_out, att, ROWS_, C_, DI_);
    }

    // 9) residual / mask fusion
    k_final<<<(B_ * C_ * L_) / 256, blk, 0, stream>>>(x, mask, conv_out, inorm,
                                                      att, out);
}
